// MambaBlock_35304631174087
// MI455X (gfx1250) — matrix operand order repaired and checked by reading, not compiled, not run on hardware
//
#include <hip/hip_runtime.h>
#include <hip/hip_bf16.h>

// ---------------------------------------------------------------------------
// Mamba block for MI455X (gfx1250): all GEMMs via v_wmma_f32_16x16x32_f16
// (32x64 wave tiles, pointer-increment addressing, no spills), selective scan
// staged through the Tensor Data Mover, recurrence in f32 VALU. wave32.
// ---------------------------------------------------------------------------

typedef __attribute__((ext_vector_type(16))) _Float16     v16h;
typedef __attribute__((ext_vector_type(8)))  float        v8f;
typedef __attribute__((ext_vector_type(4)))  unsigned int v4u;
typedef __attribute__((ext_vector_type(8)))  int          v8i;
typedef __attribute__((ext_vector_type(4)))  int          v4i;

#define D_MODEL 1024
#define D_INNER 2048
#define D_STATE 16
#define D_CONV  4
#define DT_RANK 64
#define B_SZ    2
#define L_SEQ   1024
#define M_ROWS  (B_SZ * L_SEQ)           // 2048 token rows
#define N_PROJ  (DT_RANK + 2 * D_STATE)  // 96

// ---------------------------------------------------------------------------
// WMMA fragment loaders (A and B both use contiguous k = 16*half + s; wave32).
// Pointer-based: caller advances the per-lane row pointer by 32 elems per step.
// ---------------------------------------------------------------------------

// A fragment from rowp = &A[(m0 + lane%16) * K + k0]; two b128 loads at
// byte offsets 32*half and 32*half + 16.
__device__ __forceinline__ v16h frag_a_from(const _Float16* rowp, int half) {
    union { v16h h; uint4 q[2]; } fa;
    const uint4* p = reinterpret_cast<const uint4*>(rowp + half * 16);
    fa.q[0] = p[0];   // K elems half*16 .. half*16+7
    fa.q[1] = p[1];   // K elems half*16+8 .. half*16+15
    return fa.h;
}

// B fragment from p = &W[(n0 + lane%16) * K + k0 + (lane/16)*16];
// 32 contiguous bytes per lane.
__device__ __forceinline__ v16h frag_b_from(const _Float16* p) {
    union { v16h h; uint4 q[2]; } fb;
    const uint4* p4 = reinterpret_cast<const uint4*>(p);
    fb.q[0] = p4[0];
    fb.q[1] = p4[1];
    return fb.h;
}

// C/D store: VGPR i, lane -> (M = m0 + i + (lane>=16)*8, N = n0 + lane%16)
__device__ __forceinline__ void store_frag_c(float* C, int ldc, int m0, int n0, v8f acc) {
    const int lane = threadIdx.x & 31;
    const int n    = n0 + (lane & 15);
    const int mb   = m0 + (lane >> 4) * 8;
#pragma unroll
    for (int i = 0; i < 8; ++i)
        C[(size_t)(mb + i) * ldc + n] = acc[i];
}

// ---------------------------------------------------------------------------
// GEMM: C(M,N) f32 = A(M,K) f16 x W(N,K)^T f16.  Wave tile 32x64:
// 2 A frags x 4 B frags -> 8 WMMAs per 32-deep K step.  Six per-lane row
// pointers are advanced (+32 elems) each step; prefetch uses the same
// pointers with an immediate offset (next cacheline).  Exact grid,
// M%32==0, N%64==0, K%32==0.
// ---------------------------------------------------------------------------
__global__ void __launch_bounds__(256, 1)
gemm_wmma_nt_x8(const _Float16* __restrict__ A,
                const _Float16* __restrict__ W,
                float* __restrict__ C,
                int M, int N, int K) {
    const int wave   = (blockIdx.x * blockDim.x + threadIdx.x) >> 5;
    const int ntiles = N >> 6;                 // 64-wide strips
    const int m0 = (wave / ntiles) * 32;
    const int n0 = (wave % ntiles) * 64;
    const int lane = threadIdx.x & 31;
    const int half = lane >> 4;
    const int l16  = lane & 15;
    const int koff = half * 16;

    const _Float16* pa0 = A + (size_t)(m0      + l16) * K;
    const _Float16* pa1 = A + (size_t)(m0 + 16 + l16) * K;
    const _Float16* pb0 = W + (size_t)(n0      + l16) * K + koff;
    const _Float16* pb1 = W + (size_t)(n0 + 16 + l16) * K + koff;
    const _Float16* pb2 = W + (size_t)(n0 + 32 + l16) * K + koff;
    const _Float16* pb3 = W + (size_t)(n0 + 48 + l16) * K + koff;

    v8f acc00 = {}, acc01 = {}, acc02 = {}, acc03 = {};
    v8f acc10 = {}, acc11 = {}, acc12 = {}, acc13 = {};

    for (int k0 = 0; k0 < K; k0 += 32) {
        // next-cacheline prefetch (speculative; dropped if past allocation)
        __builtin_prefetch(pa0 + 64, 0, 3);
        __builtin_prefetch(pa1 + 64, 0, 3);
        __builtin_prefetch(pb0 + 64, 0, 3);
        __builtin_prefetch(pb1 + 64, 0, 3);
        __builtin_prefetch(pb2 + 64, 0, 3);
        __builtin_prefetch(pb3 + 64, 0, 3);

        const v16h a0 = frag_a_from(pa0, half);
        const v16h a1 = frag_a_from(pa1, half);
        const v16h b0 = frag_b_from(pb0);
        const v16h b1 = frag_b_from(pb1);
        const v16h b2 = frag_b_from(pb2);
        const v16h b3 = frag_b_from(pb3);
        acc00 = __builtin_amdgcn_wmma_f32_16x16x32_f16(false, a0, false, b0, (short)0, acc00, false, false);
        acc01 = __builtin_amdgcn_wmma_f32_16x16x32_f16(false, a0, false, b1, (short)0, acc01, false, false);
        acc02 = __builtin_amdgcn_wmma_f32_16x16x32_f16(false, a0, false, b2, (short)0, acc02, false, false);
        acc03 = __builtin_amdgcn_wmma_f32_16x16x32_f16(false, a0, false, b3, (short)0, acc03, false, false);
        acc10 = __builtin_amdgcn_wmma_f32_16x16x32_f16(false, a1, false, b0, (short)0, acc10, false, false);
        acc11 = __builtin_amdgcn_wmma_f32_16x16x32_f16(false, a1, false, b1, (short)0, acc11, false, false);
        acc12 = __builtin_amdgcn_wmma_f32_16x16x32_f16(false, a1, false, b2, (short)0, acc12, false, false);
        acc13 = __builtin_amdgcn_wmma_f32_16x16x32_f16(false, a1, false, b3, (short)0, acc13, false, false);

        pa0 += 32; pa1 += 32;
        pb0 += 32; pb1 += 32; pb2 += 32; pb3 += 32;
    }
    store_frag_c(C, N, m0,      n0,      acc00);
    store_frag_c(C, N, m0,      n0 + 16, acc01);
    store_frag_c(C, N, m0,      n0 + 32, acc02);
    store_frag_c(C, N, m0,      n0 + 48, acc03);
    store_frag_c(C, N, m0 + 16, n0,      acc10);
    store_frag_c(C, N, m0 + 16, n0 + 16, acc11);
    store_frag_c(C, N, m0 + 16, n0 + 32, acc12);
    store_frag_c(C, N, m0 + 16, n0 + 48, acc13);
}

// GEMM for narrow N (x_proj: N=96). Wave tile 16x16.
__global__ void gemm_wmma_nt(const _Float16* __restrict__ A,
                             const _Float16* __restrict__ W,
                             float* __restrict__ C,
                             int M, int N, int K) {
    const int wave   = (blockIdx.x * blockDim.x + threadIdx.x) >> 5;
    const int ntiles = N >> 4;
    const int m0 = (wave / ntiles) * 16;
    const int n0 = (wave % ntiles) * 16;
    const int lane = threadIdx.x & 31;
    const int half = lane >> 4;
    const int l16  = lane & 15;

    const _Float16* pa = A + (size_t)(m0 + l16) * K;
    const _Float16* pb = W + (size_t)(n0 + l16) * K + half * 16;

    v8f acc = {};
    for (int k0 = 0; k0 < K; k0 += 32) {
        const v16h a = frag_a_from(pa, half);
        const v16h b = frag_b_from(pb);
        acc = __builtin_amdgcn_wmma_f32_16x16x32_f16(false, a, false, b, (short)0, acc, false, false);
        pa += 32; pb += 32;
    }
    store_frag_c(C, N, m0, n0, acc);
}

// ---------------------------------------------------------------------------
// Elementwise helpers
// ---------------------------------------------------------------------------
__global__ void cast_f32_to_f16(const float* __restrict__ in, _Float16* __restrict__ out, int n) {
    const int i = blockIdx.x * blockDim.x + threadIdx.x;
    if (i < n) out[i] = (_Float16)in[i];
}

// Depthwise causal conv1d over L + bias + SiLU (x_br = xz[..., :2048]).
__global__ void conv_silu(const float* __restrict__ xz,
                          const float* __restrict__ conv_w,  // (D_INNER,1,4)
                          const float* __restrict__ conv_b,
                          float* __restrict__ u, _Float16* __restrict__ uh) {
    const int i = blockIdx.x * blockDim.x + threadIdx.x;   // over B*L*D_INNER
    if (i >= B_SZ * L_SEQ * D_INNER) return;
    const int d = i % D_INNER;
    const int l = (i / D_INNER) % L_SEQ;
    const int b = i / (D_INNER * L_SEQ);
    float acc = conv_b[d];
#pragma unroll
    for (int j = 0; j < D_CONV; ++j) {
        const int ll = l - (D_CONV - 1) + j;
        if (ll >= 0)
            acc += conv_w[d * D_CONV + j] * xz[((size_t)(b * L_SEQ + ll)) * (2 * D_INNER) + d];
    }
    const float s = acc / (1.0f + __expf(-acc));  // SiLU
    u[i]  = s;
    uh[i] = (_Float16)s;
}

// delta_raw = cols 0..63 of proj(M,96), cast to f16 for the dt_proj WMMA.
__global__ void extract_draw(const float* __restrict__ proj, _Float16* __restrict__ out, int n) {
    const int i = blockIdx.x * blockDim.x + threadIdx.x;   // over M*64
    if (i >= n) return;
    const int r = i / DT_RANK, c = i % DT_RANK;
    out[i] = (_Float16)proj[r * N_PROJ + c];
}

// delta = softplus(delta_lin + dt_proj_b), in place over (M, D_INNER).
__global__ void bias_softplus(float* __restrict__ delta, const float* __restrict__ bias, int n) {
    const int i = blockIdx.x * blockDim.x + threadIdx.x;
    if (i >= n) return;
    const float x = delta[i] + bias[i % D_INNER];
    delta[i] = (x > 20.0f) ? x : log1pf(__expf(x));
}

// ---------------------------------------------------------------------------
// TDM helper: 2D tile load (f32 elements) global -> LDS.
// Descriptor packed per CDNA5 ISA 8.3/8.4 (count=1, type=2, data_size=4B).
// ---------------------------------------------------------------------------
__device__ __forceinline__ void tdm_load_2d_f32(unsigned lds_off, const float* src,
                                                unsigned cols, unsigned rows,
                                                unsigned row_stride) {
    const unsigned long long ga = (unsigned long long)(size_t)src;
    v4u g0;
    g0[0] = 1u;                                            // count=1 (valid user D#)
    g0[1] = lds_off;                                       // lds_addr (bytes)
    g0[2] = (unsigned)(ga & 0xffffffffu);                  // global_addr[31:0]
    g0[3] = (unsigned)((ga >> 32) & 0x01ffffffu) | (2u << 30); // addr[56:32] | type=2
    v8i g1;
    g1[0] = (int)(2u << 16);                               // data_size = 4 bytes
    g1[1] = (int)((cols & 0xffffu) << 16);                 // tensor_dim0[15:0]
    g1[2] = (int)((cols >> 16) | ((rows & 0xffffu) << 16));// dim0[31:16] | dim1[15:0]
    g1[3] = (int)((rows >> 16) | ((cols & 0xffffu) << 16));// dim1[31:16] | tile_dim0
    g1[4] = (int)(rows & 0xffffu);                         // tile_dim1 (tile_dim2=0)
    g1[5] = (int)row_stride;                               // tensor_dim0_stride[31:0]
    g1[6] = 0;                                             // stride0 hi | stride1 lo
    g1[7] = 0;                                             // stride1 hi
    v4i gz = {0, 0, 0, 0};
#if __clang_major__ >= 23
    v8i gz8 = {0, 0, 0, 0, 0, 0, 0, 0};
    __builtin_amdgcn_tensor_load_to_lds(g0, g1, gz, gz, gz8, 0);
#else
    __builtin_amdgcn_tensor_load_to_lds(g0, g1, gz, gz, 0);
#endif
}

// ---------------------------------------------------------------------------
// Selective scan: one lane per (b,d) channel, h[16] in VGPRs.  B_t/C_t for
// 128 timesteps staged into LDS by the Tensor Data Mover (32x128 f32 tile,
// tensor row stride 96), one barrier pair per chunk instead of per step.
// SiLU(z) gate fused; emits f16 for the out_proj WMMA.
// grid = (D_INNER/256, B_SZ), block = 256.
// ---------------------------------------------------------------------------
#define SCAN_CHUNK 128

__global__ void selective_scan(const float* __restrict__ delta,   // (B,L,DI)
                               const float* __restrict__ u,       // (B,L,DI)
                               const float* __restrict__ proj,    // (B,L,96): [64..79]=B, [80..95]=C
                               const float* __restrict__ xz,      // (B,L,2*DI): z at +DI
                               const float* __restrict__ A_log,   // (DI,16)
                               const float* __restrict__ D_param, // (DI)
                               _Float16* __restrict__ yg) {       // (B,L,DI)
    const int b = blockIdx.y;
    const int d = blockIdx.x * blockDim.x + threadIdx.x;

    __shared__ float sBC[SCAN_CHUNK * 32];   // [t][0..15]=B_t, [t][16..31]=C_t

    float Areg[D_STATE], h[D_STATE];
#pragma unroll
    for (int s = 0; s < D_STATE; ++s) {
        Areg[s] = -__expf(A_log[d * D_STATE + s]);
        h[s] = 0.0f;
    }
    const float Dp = D_param[d];

    for (int c0 = 0; c0 < L_SEQ; c0 += SCAN_CHUNK) {
        __syncthreads();                      // previous chunk fully consumed
        if (threadIdx.x < 32) {               // wave 0 drives the TDM
            const size_t row0 = (size_t)(b * L_SEQ + c0);
            tdm_load_2d_f32((unsigned)(size_t)(void*)sBC,
                            proj + row0 * N_PROJ + DT_RANK,
                            /*cols=*/32, /*rows=*/SCAN_CHUNK, /*row_stride=*/N_PROJ);
            __builtin_amdgcn_s_wait_tensorcnt(0);
        }
        __syncthreads();                      // tile visible to all 8 waves

        for (int t = 0; t < SCAN_CHUNK; ++t) {
            const size_t row = (size_t)(b * L_SEQ + c0 + t);
            const size_t idx = row * D_INNER + d;
            const float* bc  = &sBC[t * 32];
            const float dt = delta[idx];
            const float ut = u[idx];
            const float du = dt * ut;
            float y = Dp * ut;
#pragma unroll
            for (int s = 0; s < D_STATE; ++s) {
                h[s] = __expf(dt * Areg[s]) * h[s] + du * bc[s];
                y += h[s] * bc[16 + s];
            }
            const float zt = xz[row * (2 * D_INNER) + D_INNER + d];
            const float g  = zt / (1.0f + __expf(-zt));    // SiLU gate
            yg[idx] = (_Float16)(y * g);
        }
    }
}

// ---------------------------------------------------------------------------
// Host-side launcher
// ---------------------------------------------------------------------------
static inline size_t align256(size_t x) { return (x + 255) & ~(size_t)255; }

extern "C" void kernel_launch(void* const* d_in, const int* in_sizes, int n_in,
                              void* d_out, int out_size, void* d_ws, size_t ws_size,
                              hipStream_t stream) {
    (void)in_sizes; (void)n_in; (void)out_size; (void)ws_size;
    const float* x         = (const float*)d_in[0];
    const float* in_proj_w = (const float*)d_in[1];
    const float* conv_w    = (const float*)d_in[2];
    const float* conv_b    = (const float*)d_in[3];
    const float* x_proj_w  = (const float*)d_in[4];
    const float* dt_proj_w = (const float*)d_in[5];
    const float* dt_proj_b = (const float*)d_in[6];
    const float* A_log     = (const float*)d_in[7];
    const float* D_param   = (const float*)d_in[8];
    const float* out_proj_w= (const float*)d_in[9];
    float* out = (float*)d_out;

    // ---- workspace carve-out ----
    char* ws = (char*)d_ws;
    size_t off = 0;
    auto carve = [&](size_t bytes) { char* p = ws + off; off += align256(bytes); return p; };

    _Float16* xh    = (_Float16*)carve((size_t)M_ROWS * D_MODEL * 2);
    _Float16* w_in  = (_Float16*)carve((size_t)2 * D_INNER * D_MODEL * 2);
    _Float16* w_xp  = (_Float16*)carve((size_t)N_PROJ * D_INNER * 2);
    _Float16* w_dt  = (_Float16*)carve((size_t)D_INNER * DT_RANK * 2);
    _Float16* w_out = (_Float16*)carve((size_t)D_MODEL * D_INNER * 2);
    float*    xz    = (float*)   carve((size_t)M_ROWS * 2 * D_INNER * 4);
    float*    u     = (float*)   carve((size_t)M_ROWS * D_INNER * 4);
    _Float16* uh    = (_Float16*)carve((size_t)M_ROWS * D_INNER * 2);
    float*    proj  = (float*)   carve((size_t)M_ROWS * N_PROJ * 4);
    _Float16* drawh = (_Float16*)carve((size_t)M_ROWS * DT_RANK * 2);
    float*    delta = (float*)   carve((size_t)M_ROWS * D_INNER * 4);
    _Float16* yg    = (_Float16*)carve((size_t)M_ROWS * D_INNER * 2);

    const int TB = 256;
    auto blocks = [](int n, int tb) { return (n + tb - 1) / tb; };

    // 1) casts to f16
    { int n = M_ROWS * D_MODEL;      cast_f32_to_f16<<<blocks(n,TB),TB,0,stream>>>(x, xh, n); }
    { int n = 2 * D_INNER * D_MODEL; cast_f32_to_f16<<<blocks(n,TB),TB,0,stream>>>(in_proj_w, w_in, n); }
    { int n = N_PROJ * D_INNER;      cast_f32_to_f16<<<blocks(n,TB),TB,0,stream>>>(x_proj_w, w_xp, n); }
    { int n = D_INNER * DT_RANK;     cast_f32_to_f16<<<blocks(n,TB),TB,0,stream>>>(dt_proj_w, w_dt, n); }
    { int n = D_MODEL * D_INNER;     cast_f32_to_f16<<<blocks(n,TB),TB,0,stream>>>(out_proj_w, w_out, n); }

    // 2) in_proj: xz(2048,4096) = xh(2048,1024) . w_in(4096,1024)^T
    { int waves = (M_ROWS/32) * ((2*D_INNER)/64);
      gemm_wmma_nt_x8<<<waves*32/TB, TB, 0, stream>>>(xh, w_in, xz, M_ROWS, 2*D_INNER, D_MODEL); }

    // 3) depthwise conv + SiLU -> u, uh
    { int n = B_SZ * L_SEQ * D_INNER;
      conv_silu<<<blocks(n,TB),TB,0,stream>>>(xz, conv_w, conv_b, u, uh); }

    // 4) x_proj: proj(2048,96) = uh(2048,2048) . w_xp(96,2048)^T
    { int waves = (M_ROWS/16) * (N_PROJ/16);
      gemm_wmma_nt<<<waves*32/TB, TB, 0, stream>>>(uh, w_xp, proj, M_ROWS, N_PROJ, D_INNER); }

    // 5) dt_proj: delta(2048,2048) = drawh(2048,64) . w_dt(2048,64)^T, then softplus
    { int n = M_ROWS * DT_RANK;
      extract_draw<<<blocks(n,TB),TB,0,stream>>>(proj, drawh, n); }
    { int waves = (M_ROWS/32) * (D_INNER/64);
      gemm_wmma_nt_x8<<<waves*32/TB, TB, 0, stream>>>(drawh, w_dt, delta, M_ROWS, D_INNER, DT_RANK); }
    { int n = M_ROWS * D_INNER;
      bias_softplus<<<blocks(n,TB),TB,0,stream>>>(delta, dt_proj_b, n); }

    // 6) selective scan (TDM-staged B/C) + SiLU(z) gate -> yg (f16)
    { dim3 grid(D_INNER / TB, B_SZ);
      selective_scan<<<grid, TB, 0, stream>>>(delta, u, proj, xz, A_log, D_param, yg); }

    // 7) out_proj: out(2048,1024) = yg(2048,2048) . w_out(1024,2048)^T
    { int waves = (M_ROWS/32) * (D_MODEL/64);
      gemm_wmma_nt_x8<<<waves*32/TB, TB, 0, stream>>>(yg, w_out, out, M_ROWS, D_MODEL, D_INNER); }
}
